// MultiHeadedAttention_81235011437255
// MI455X (gfx1250) — matrix operand order repaired and checked by reading, not compiled, not run on hardware
//
#include <hip/hip_runtime.h>
#include <hip/hip_bf16.h>
#include <cstdint>
#include <cstddef>

// Problem constants (reference: B=2, S=2048, D=1024, H=16, DH=64)
#define BB   2
#define SSEQ 2048
#define DMOD 1024
#define NH   16
#define DHD  64
#define BSROWS (BB * SSEQ)          // 4096 token rows
#define KC   256                    // ctx-GEMM K chunk staged in LDS (64 x 256 bf16 = 32 KB)

typedef __bf16 bf16_t;
typedef __attribute__((ext_vector_type(8)))  __bf16 v8bf;
typedef __attribute__((ext_vector_type(16))) __bf16 v16bf;
typedef __attribute__((ext_vector_type(8)))  float  v8f;
typedef __attribute__((ext_vector_type(4)))  unsigned int v4u;
typedef __attribute__((ext_vector_type(8)))  int v8i;
typedef __attribute__((ext_vector_type(4)))  int v4i;

// ---------- helpers ----------

__device__ __forceinline__ bf16_t f2bf(float x) {
  uint32_t u = __builtin_bit_cast(uint32_t, x);
  u += 0x7FFFu + ((u >> 16) & 1u);            // round-to-nearest-even
  uint16_t h = (uint16_t)(u >> 16);
  return __builtin_bit_cast(bf16_t, h);
}

__device__ __forceinline__ v8f wmma_bf16(v16bf a, v16bf b, v8f c) {
  // D(f32 16x16) = A(16x32 bf16) * B(32x16 bf16) + C
  return __builtin_amdgcn_wmma_f32_16x16x32_bf16(
      /*neg_a=*/false, a, /*neg_b=*/false, b,
      /*c_mod=*/(short)0, c, /*reuse_a=*/false, /*reuse_b=*/false);
}

// A-fragment: row-major bf16 source. rowk points at (row, kk). contiguous k order used here:
// lane half=lane>>4: elems 0..15 <- K = kk+16*half+(0..15)
__device__ __forceinline__ v16bf load_a_frag(const bf16_t* __restrict__ rowk, int half) {
  v8bf x0 = *reinterpret_cast<const v8bf*>(rowk + 16 * half);
  v8bf x1 = *reinterpret_cast<const v8bf*>(rowk + 16 * half + 8);
  v16bf a;
#pragma unroll
  for (int i = 0; i < 8; ++i) { a[i] = x0[i]; a[8 + i] = x1[i]; }
  return a;
}

// A-fragment sourced from f32 (attn probabilities), converted to bf16 on the fly.
__device__ __forceinline__ v16bf load_a_frag_f32(const float* __restrict__ rowk, int half) {
  v8f x0 = *reinterpret_cast<const v8f*>(rowk + 16 * half);
  v8f x1 = *reinterpret_cast<const v8f*>(rowk + 16 * half + 8);
  v16bf a;
#pragma unroll
  for (int i = 0; i < 8; ++i) { a[i] = f2bf(x0[i]); a[8 + i] = f2bf(x1[i]); }
  return a;
}

// B-fragment: lane holds column n = lane&15, 16 contiguous K values starting at
// kk + 16*(lane>>4). p points at that run (source stored K-contiguous per column).
// Works for both global and LDS (generic) pointers.
__device__ __forceinline__ v16bf load_b_frag(const bf16_t* p) {
  v8bf x0 = *reinterpret_cast<const v8bf*>(p);
  v8bf x1 = *reinterpret_cast<const v8bf*>(p + 8);
  v16bf b;
#pragma unroll
  for (int i = 0; i < 8; ++i) { b[i] = x0[i]; b[8 + i] = x1[i]; }
  return b;
}

__device__ __forceinline__ v8f zero_v8f() {
  v8f z;
#pragma unroll
  for (int i = 0; i < 8; ++i) z[i] = 0.0f;
  return z;
}

// ---------- elementwise prep ----------

__global__ void mha_cast_bf16(const float* __restrict__ src, bf16_t* __restrict__ dst, int n) {
  int i = blockIdx.x * blockDim.x + threadIdx.x;
  if (i < n) dst[i] = f2bf(src[i]);
}

// Wt[n][k] = W[k][n] with f32->bf16 cast (weights become K-contiguous per output column)
__global__ void mha_transpose_cast(const float* __restrict__ W, bf16_t* __restrict__ Wt) {
  int i = blockIdx.x * blockDim.x + threadIdx.x;
  if (i >= DMOD * DMOD) return;
  int n = i / DMOD, k = i % DMOD;
  Wt[i] = f2bf(W[k * DMOD + n]);
}

// ---------- projection GEMM (register-blocked 16x64 per wave) ----------
// P = X[4096x1024] * W + bias.  mode 0: Q (scaled 1/8) -> [B,H,S,DH]
// mode 1: K -> [B,H,S,DH]      mode 2: V -> [B,H,DH,S] (transposed)
__global__ void mha_proj_gemm(const bf16_t* __restrict__ X, const bf16_t* __restrict__ Wt,
                              const float* __restrict__ bias, bf16_t* __restrict__ out,
                              int mode) {
  const int wave = threadIdx.x >> 5;
  const int lane = threadIdx.x & 31;
  const int half = lane >> 4;
  const int ln   = lane & 15;
  const int tile = blockIdx.x * 8 + wave;          // 256 * 16 = 4096 strips
  const int tM  = tile / (DMOD / 64);              // 0..255
  const int tNq = tile % (DMOD / 64);              // 0..15 (64-wide N strip)

  v8f acc[4];
#pragma unroll
  for (int j = 0; j < 4; ++j) acc[j] = zero_v8f();

  const bf16_t* arow_base = X + (size_t)(tM * 16 + ln) * DMOD;
  const bf16_t* b_base    = Wt + (size_t)(tNq * 64 + ln) * DMOD + 16 * half;
#pragma unroll 2
  for (int kk = 0; kk < DMOD; kk += 32) {
    v16bf a = load_a_frag(arow_base + kk, half);
#pragma unroll
    for (int j = 0; j < 4; ++j) {
      v16bf b = load_b_frag(b_base + (size_t)(j * 16) * DMOD + kk);
      acc[j] = wmma_bf16(a, b, acc[j]);
    }
  }

#pragma unroll
  for (int j = 0; j < 4; ++j) {
    const int col = tNq * 64 + j * 16 + ln;
    const float bv = bias[col];
    const int h = col >> 6;            // col / DH
    const int dd = col & 63;           // col % DH
#pragma unroll
    for (int v = 0; v < 8; ++v) {
      int row = tM * 16 + v + 8 * half;
      int bb = row >> 11;              // row / S
      int s  = row & (SSEQ - 1);       // row % S
      float val = acc[j][v] + bv;
      if (mode == 0) val *= 0.125f;
      size_t addr;
      if (mode == 2) addr = (((size_t)(bb * NH + h) * DHD + dd) * SSEQ + s);
      else           addr = (((size_t)(bb * NH + h) * SSEQ + s) * DHD + dd);
      out[addr] = f2bf(val);
    }
  }
}

// ---------- symmetric scores (16x64 per wave; 2 A-frags feed 8 WMMAs/k-step) ----------
// S[I,Jq] = 0.5 * (Q_I K_J^T + K_I Q_J^T) for 4 J subtiles; Q carries 1/sqrt(DH).
__global__ void mha_scores(const bf16_t* __restrict__ Qb, const bf16_t* __restrict__ Kb,
                           float* __restrict__ attn) {
  const int wave = threadIdx.x >> 5;
  const int lane = threadIdx.x & 31;
  const int half = lane >> 4;
  const int ln   = lane & 15;
  const int tile = blockIdx.x * 8 + wave;          // B*H * 128 * 32 = 131072 strips
  const int TPH  = (SSEQ / 16) * (SSEQ / 64);      // 128 * 32 = 4096
  const int bh = tile / TPH;
  const int r  = tile % TPH;
  const int tI  = r / (SSEQ / 64);                 // 0..127
  const int tJq = r % (SSEQ / 64);                 // 0..31 (64-wide J strip)

  const bf16_t* Qh = Qb + (size_t)bh * SSEQ * DHD;
  const bf16_t* Kh = Kb + (size_t)bh * SSEQ * DHD;

  v8f acc1[4], acc2[4];
#pragma unroll
  for (int j = 0; j < 4; ++j) { acc1[j] = zero_v8f(); acc2[j] = zero_v8f(); }

  const bf16_t* qi = Qh + (size_t)(tI * 16 + ln) * DHD;
  const bf16_t* ki = Kh + (size_t)(tI * 16 + ln) * DHD;
#pragma unroll
  for (int kk = 0; kk < DHD; kk += 32) {
    v16bf aQ = load_a_frag(qi + kk, half);
    v16bf aK = load_a_frag(ki + kk, half);
#pragma unroll
    for (int j = 0; j < 4; ++j) {
      const size_t jrow = (size_t)(tJq * 64 + j * 16 + ln) * DHD + kk + 16 * half;
      v16bf bK = load_b_frag(Kh + jrow);
      acc1[j] = wmma_bf16(aQ, bK, acc1[j]);
      v16bf bQ = load_b_frag(Qh + jrow);
      acc2[j] = wmma_bf16(aK, bQ, acc2[j]);
    }
  }

  float* Sout = attn + (size_t)bh * SSEQ * SSEQ;
#pragma unroll
  for (int j = 0; j < 4; ++j) {
    const int col = tJq * 64 + j * 16 + ln;
#pragma unroll
    for (int v = 0; v < 8; ++v) {
      int row = tI * 16 + v + 8 * half;
      Sout[(size_t)row * SSEQ + col] = 0.5f * (acc1[j][v] + acc2[j][v]);
    }
  }
}

// ---------- row softmax (in place) ----------
__global__ void mha_softmax(float* __restrict__ attn) {
  __shared__ float red[256];
  float* p = attn + (size_t)blockIdx.x * SSEQ;
  const int tid = threadIdx.x;

  float lmax = -3.4e38f;
  for (int i = tid; i < SSEQ; i += 256) lmax = fmaxf(lmax, p[i]);
  red[tid] = lmax; __syncthreads();
  for (int s = 128; s > 0; s >>= 1) { if (tid < s) red[tid] = fmaxf(red[tid], red[tid + s]); __syncthreads(); }
  const float m = red[0]; __syncthreads();

  float lsum = 0.0f;
  for (int i = tid; i < SSEQ; i += 256) { float e = __expf(p[i] - m); p[i] = e; lsum += e; }
  red[tid] = lsum; __syncthreads();
  for (int s = 128; s > 0; s >>= 1) { if (tid < s) red[tid] += red[tid + s]; __syncthreads(); }
  const float inv = 1.0f / red[0];

  for (int i = tid; i < SSEQ; i += 256) p[i] *= inv;
}

// ---------- post-softmax symmetrization (in place, one thread per i<j pair) ----------
__global__ void mha_symmetrize(float* __restrict__ attn) {
  size_t idx = (size_t)blockIdx.x * 256 + threadIdx.x;     // over B*H*S*S
  int col = (int)(idx & (SSEQ - 1));
  size_t t = idx >> 11;
  int row = (int)(t & (SSEQ - 1));
  size_t bh = t >> 11;
  if (row < col) {
    float* P = attn + bh * (size_t)SSEQ * SSEQ;
    float a = P[(size_t)row * SSEQ + col];
    float b = P[(size_t)col * SSEQ + row];
    float m = 0.5f * (a + b);
    P[(size_t)row * SSEQ + col] = m;
    P[(size_t)col * SSEQ + row] = m;
  }
}

// ---------- ctx = attn @ V, V^T chunk staged in LDS via Tensor Data Mover ----------
// Block = 8 waves; wave w computes rows [m0 + 16w, +16) x all 64 cols of one head.
// V^T pane [64][S] is consumed in [64][KC] LDS chunks loaded by the TDM.
__global__ void mha_ctx_gemm(const float* __restrict__ attn, const bf16_t* __restrict__ Vtb,
                             bf16_t* __restrict__ ctx) {
  __shared__ bf16_t sV[DHD * KC];   // 64 x 256 bf16 = 32 KB; ONLY shared object in this
                                    // kernel -> static LDS offset 0 (used as D#.lds_addr)
  const int wave = threadIdx.x >> 5;
  const int lane = threadIdx.x & 31;
  const int half = lane >> 4;
  const int ln   = lane & 15;
  const int bh     = blockIdx.x >> 4;     // 0..31
  const int mchunk = blockIdx.x & 15;     // 16 chunks of 128 rows
  const int mrow0  = mchunk * 128 + wave * 16;

  const float*  A  = attn + (size_t)bh * SSEQ * SSEQ;
  const bf16_t* Vt = Vtb + (size_t)bh * DHD * SSEQ;

  v8f acc[4];
#pragma unroll
  for (int j = 0; j < 4; ++j) acc[j] = zero_v8f();

  for (int k0 = 0; k0 < SSEQ; k0 += KC) {
    __syncthreads();                       // previous chunk fully consumed
    if (threadIdx.x < 32) {                // wave 0 issues the TDM (uniform descriptor)
      const uint64_t ga = (uint64_t)(uintptr_t)(Vt + k0);   // tile start, bytes
      // D# group 0: count=1 | lds_addr | global_addr[56:0] | type=2
      v4u g0;
      g0[0] = 1u;                                           // count=1, user mode
      g0[1] = 0u;                                           // lds_addr = offset of sV
      g0[2] = (unsigned int)(ga & 0xFFFFFFFFu);
      g0[3] = (unsigned int)((ga >> 32) & 0x1FFFFFFu) | (2u << 30);
      // D# group 1: data_size=2B, tensor 2048x64, tile KCx64, stride0=2048
      v8i g1;
      g1[0] = (int)(1u << 16);                              // data_size code 1 (2 bytes)
      g1[1] = (int)((2048u & 0xFFFFu) << 16);               // tensor_dim0 lo
      g1[2] = (int)(((2048u >> 16) & 0xFFFFu) | (64u << 16));   // dim0 hi | tensor_dim1 lo
      g1[3] = (int)((unsigned)KC << 16);                    // tensor_dim1 hi | tile_dim0
      g1[4] = 64;                                           // tile_dim1 | tile_dim2=0
      g1[5] = 2048;                                         // tensor_dim0_stride lo
      g1[6] = 0;
      g1[7] = 0;
      v4i gz;
      gz[0] = 0; gz[1] = 0; gz[2] = 0; gz[3] = 0;
      v8i gz8;
#pragma unroll
      for (int i = 0; i < 8; ++i) gz8[i] = 0;
      // amdgpu-toolchain (clang-23) 6-arg form: (g0, g1, g2, g3, g4, cpol)
      __builtin_amdgcn_tensor_load_to_lds(g0, g1, gz, gz, gz8, 0);
      __builtin_amdgcn_s_wait_tensorcnt(0);
    }
    __syncthreads();

    const float* arow = A + (size_t)(mrow0 + ln) * SSEQ + k0;
#pragma unroll 2
    for (int kk = 0; kk < KC; kk += 32) {
      v16bf a = load_a_frag_f32(arow + kk, half);
#pragma unroll
      for (int j = 0; j < 4; ++j) {
        v16bf b = load_b_frag(sV + (j * 16 + ln) * KC + kk + 16 * half);
        acc[j] = wmma_bf16(a, b, acc[j]);
      }
    }
  }

  const int bb2 = bh / NH, h = bh % NH;
#pragma unroll
  for (int j = 0; j < 4; ++j) {
    const int dcol = j * 16 + ln;
#pragma unroll
    for (int v = 0; v < 8; ++v) {
      int s = mrow0 + v + 8 * half;
      ctx[((size_t)(bb2 * SSEQ + s)) * DMOD + h * DHD + dcol] = f2bf(acc[j][v]);
    }
  }
}

// ---------- output GEMM (register-blocked 16x64 per wave) ----------
__global__ void mha_out_gemm(const bf16_t* __restrict__ X, const bf16_t* __restrict__ Wt,
                             const float* __restrict__ bias, float* __restrict__ out) {
  const int wave = threadIdx.x >> 5;
  const int lane = threadIdx.x & 31;
  const int half = lane >> 4;
  const int ln   = lane & 15;
  const int tile = blockIdx.x * 8 + wave;
  const int tM  = tile / (DMOD / 64);
  const int tNq = tile % (DMOD / 64);

  v8f acc[4];
#pragma unroll
  for (int j = 0; j < 4; ++j) acc[j] = zero_v8f();

  const bf16_t* arow_base = X + (size_t)(tM * 16 + ln) * DMOD;
  const bf16_t* b_base    = Wt + (size_t)(tNq * 64 + ln) * DMOD + 16 * half;
#pragma unroll 2
  for (int kk = 0; kk < DMOD; kk += 32) {
    v16bf a = load_a_frag(arow_base + kk, half);
#pragma unroll
    for (int j = 0; j < 4; ++j) {
      v16bf b = load_b_frag(b_base + (size_t)(j * 16) * DMOD + kk);
      acc[j] = wmma_bf16(a, b, acc[j]);
    }
  }

#pragma unroll
  for (int j = 0; j < 4; ++j) {
    const int col = tNq * 64 + j * 16 + ln;
    const float bv = bias[col];
#pragma unroll
    for (int v = 0; v < 8; ++v) {
      int row = tM * 16 + v + 8 * half;
      out[(size_t)row * DMOD + col] = acc[j][v] + bv;
    }
  }
}

// ---------- launch ----------

extern "C" void kernel_launch(void* const* d_in, const int* in_sizes, int n_in,
                              void* d_out, int out_size, void* d_ws, size_t ws_size,
                              hipStream_t stream) {
  // setup_inputs order: key, value, query, Wq, bq, Wk, bk, Wv, bv, Wo, bo
  const float* key   = (const float*)d_in[0];
  const float* value = (const float*)d_in[1];
  const float* query = (const float*)d_in[2];
  const float* Wq = (const float*)d_in[3];  const float* bq = (const float*)d_in[4];
  const float* Wk = (const float*)d_in[5];  const float* bk = (const float*)d_in[6];
  const float* Wv = (const float*)d_in[7];  const float* bv = (const float*)d_in[8];
  const float* Wo = (const float*)d_in[9];  const float* bo = (const float*)d_in[10];

  float* out_proj = (float*)d_out;                                  // [B,S,D]
  float* attn     = (float*)d_out + (size_t)BSROWS * DMOD;          // [B,H,S,S]

  // workspace carve-up (bf16), 64 MB total
  const size_t XSZ = (size_t)BSROWS * DMOD;      // 4,194,304 elems
  const size_t WSZ = (size_t)DMOD * DMOD;        // 1,048,576 elems
  bf16_t* p = (bf16_t*)d_ws;
  bf16_t* Xq  = p;  p += XSZ;
  bf16_t* Xk  = p;  p += XSZ;
  bf16_t* Xv  = p;  p += XSZ;
  bf16_t* WtQ = p;  p += WSZ;
  bf16_t* WtK = p;  p += WSZ;
  bf16_t* WtV = p;  p += WSZ;
  bf16_t* WtO = p;  p += WSZ;
  bf16_t* Qb  = p;  p += XSZ;   // [B,H,S,DH]
  bf16_t* Kb  = p;  p += XSZ;   // [B,H,S,DH]
  bf16_t* Vtb = p;  p += XSZ;   // [B,H,DH,S]
  bf16_t* Ctx = p;  p += XSZ;   // [B,S,D]

  const int threads = 256;

  // 1) casts
  mha_cast_bf16<<<(int)(XSZ / threads), threads, 0, stream>>>(query, Xq, (int)XSZ);
  mha_cast_bf16<<<(int)(XSZ / threads), threads, 0, stream>>>(key,   Xk, (int)XSZ);
  mha_cast_bf16<<<(int)(XSZ / threads), threads, 0, stream>>>(value, Xv, (int)XSZ);
  mha_transpose_cast<<<(int)(WSZ / threads), threads, 0, stream>>>(Wq, WtQ);
  mha_transpose_cast<<<(int)(WSZ / threads), threads, 0, stream>>>(Wk, WtK);
  mha_transpose_cast<<<(int)(WSZ / threads), threads, 0, stream>>>(Wv, WtV);
  mha_transpose_cast<<<(int)(WSZ / threads), threads, 0, stream>>>(Wo, WtO);

  // 2) projections: 4096 strips / 8 waves = 512 blocks
  mha_proj_gemm<<<512, threads, 0, stream>>>(Xq, WtQ, bq, Qb, 0);
  mha_proj_gemm<<<512, threads, 0, stream>>>(Xk, WtK, bk, Kb, 1);
  mha_proj_gemm<<<512, threads, 0, stream>>>(Xv, WtV, bv, Vtb, 2);

  // 3) symmetric scores: 131072 strips / 8 waves = 16384 blocks
  mha_scores<<<16384, threads, 0, stream>>>(Qb, Kb, attn);

  // 4) row softmax in place (B*H*S = 65536 rows)
  mha_softmax<<<BB * NH * SSEQ, threads, 0, stream>>>(attn);

  // 5) post-softmax symmetrize in place (B*H*S*S threads)
  mha_symmetrize<<<(unsigned)(((size_t)BB * NH * SSEQ * SSEQ) / threads), threads, 0, stream>>>(attn);

  // 6) ctx = attn @ V with TDM-staged LDS: 32 heads * 16 row-chunks = 512 blocks
  mha_ctx_gemm<<<512, threads, 0, stream>>>(attn, Vtb, Ctx);

  // 7) output projection: 512 blocks
  mha_out_gemm<<<512, threads, 0, stream>>>(Ctx, WtO, bo, out_proj);
}
